// SignatureAugmentedAttention_58884001628524
// MI455X (gfx1250) — matrix operand order repaired and checked by reading, not compiled, not run on hardware
//
#include <hip/hip_runtime.h>
#include <hip/hip_bf16.h>

typedef _Float16 v16h __attribute__((ext_vector_type(16)));
typedef _Float16 v8h  __attribute__((ext_vector_type(8)));
typedef float    v8f  __attribute__((ext_vector_type(8)));

#define BATCH 16
#define SEQ   1024
#define DMODEL 1024
#define NH    16
#define HDK   64

static __device__ __forceinline__ v8f wmma16(v16h a, v16h b, v8f c) {
    // D = A(16x32 f16) * B(32x16 f16) + C(16x16 f32)
    return __builtin_amdgcn_wmma_f32_16x16x32_f16(false, a, false, b, (short)0, c, false, false);
}

// One 16-byte async copy global -> LDS (ASYNCcnt-tracked).
static __device__ __forceinline__ void async_copy_b128(unsigned lds_addr, unsigned long long gaddr) {
    asm volatile("global_load_async_to_lds_b128 %0, %1, off"
                 :: "v"(lds_addr), "v"(gaddr) : "memory");
}

// A fragment (16x32, MxK) from row-major fp16, row stride ld (halves).
// Layout used here: lane m=L%16, g=L/16; halves 0..15 -> K=k0+16g+0..15 (contiguous)
static __device__ __forceinline__ v16h load_a_frag(const _Float16* __restrict__ base,
                                                   long ld, int row0, int k0, int lane) {
    int m = lane & 15, g = lane >> 4;
    const _Float16* p = base + (long)(row0 + m) * ld + k0 + 16 * g;
    union { v16h v; v8h h[2]; } u;
    u.h[0] = *(const v8h*)(p);
    u.h[1] = *(const v8h*)(p + 8);
    return u.v;
}

// B fragment (32x16, KxN) from "Bt" storage laid out [N][K] row-major (stride ld halves).
// Layout used here: lane n=L%16, g=L/16; halves 0..15 -> K=k0+16g+0..15 (contiguous)
static __device__ __forceinline__ v16h load_b_frag(const _Float16* __restrict__ base,
                                                   long ld, int n0, int k0, int lane) {
    int n = lane & 15, g = lane >> 4;
    return *(const v16h*)(base + (long)(n0 + n) * ld + k0 + 16 * g);
}

// ---------------------------------------------------------------- converts
__global__ void __launch_bounds__(256) f32_to_f16_kernel(const float* __restrict__ src,
                                                         _Float16* __restrict__ dst, long n) {
    long i = (long)blockIdx.x * blockDim.x + threadIdx.x;
    if (i < n) dst[i] = (_Float16)src[i];
}

// dst[n][k] = (f16) src[k][n] ; src is rows x cols (K x N), dst is cols x rows
__global__ void __launch_bounds__(256) transpose_f32_to_f16_kernel(const float* __restrict__ src,
                                                                   _Float16* __restrict__ dst,
                                                                   int rows, int cols) {
    long idx = (long)blockIdx.x * blockDim.x + threadIdx.x;
    long total = (long)rows * cols;
    if (idx >= total) return;
    int n = (int)(idx % cols);
    int k = (int)(idx / cols);
    dst[(long)n * rows + k] = (_Float16)src[idx];
}

// ---------------------------------------------------------------- WMMA GEMM
// C[M,N] = A[M,K](f16,row-major) * Bt[N,K](f16,"transposed B")
// B tile (64 cols x 32 k) is staged in LDS via async copies, double-buffered,
// shared by all 8 waves (removes the 8x redundant global B traffic).
// mode 0: f16 out, layout [b][h][s][dk]   (Q, K projections)
// mode 1: f16 out, layout [b][h][dk][s]   (V projection, pre-transposed for attn*V)
// mode 2: f32 out, row-major [M][N]       (final output projection)
__global__ void __launch_bounds__(256) gemm_wmma_kernel(const _Float16* __restrict__ A,
                                                        const _Float16* __restrict__ Bt,
                                                        void* __restrict__ Cv,
                                                        int M, int N, int K, int mode) {
    __shared__ _Float16 bs[2][64 * 32];   // two 4 KB B tiles: [n_local 0..63][k_local 0..31]

    const int tid  = threadIdx.x;
    const int lane = tid & 31;
    const int wave = tid >> 5;
    const int m0 = blockIdx.y * 256 + wave * 32;
    const int n0 = blockIdx.x * 64;
    const int nlo = lane & 15, gh = lane >> 4;

    // async-copier mapping: 256 threads x 16 B = one full 4 KB tile per k-step
    const int cp_row  = tid >> 2;        // 0..63 : local n
    const int cp_colb = (tid & 3) * 16;  // byte offset within 64 B row
    const unsigned long long gsrc =
        (unsigned long long)(const char*)(Bt + (long)(n0 + cp_row) * K) + cp_colb;
    const unsigned lds_dst0 = (unsigned)(size_t)(&bs[0][cp_row * 32]) + cp_colb;
    const unsigned lds_dst1 = (unsigned)(size_t)(&bs[1][cp_row * 32]) + cp_colb;

    v8f acc[2][4];
#pragma unroll
    for (int i = 0; i < 2; i++)
#pragma unroll
        for (int t = 0; t < 4; t++) acc[i][t] = (v8f){0.f,0.f,0.f,0.f,0.f,0.f,0.f,0.f};

    const int nsteps = K >> 5;
    async_copy_b128(lds_dst0, gsrc);           // tile 0
    async_copy_b128(lds_dst1, gsrc + 64);      // tile 1 (ks=32 -> 64 bytes)

    for (int j = 0; j < nsteps; j++) {
        const int cur = j & 1;
        if (j + 1 < nsteps) asm volatile("s_wait_asynccnt 0x1" ::: "memory");
        else                asm volatile("s_wait_asynccnt 0x0" ::: "memory");
        __syncthreads();

        v16h a0 = load_a_frag(A, K, m0, j * 32, lane);
        v16h a1 = load_a_frag(A, K, m0 + 16, j * 32, lane);
        const _Float16* bbase = &bs[cur][0];
#pragma unroll
        for (int t = 0; t < 4; t++) {
            v16h b = *(const v16h*)(bbase + (16 * t + nlo) * 32 + 16 * gh);
            acc[0][t] = wmma16(a0, b, acc[0][t]);
            acc[1][t] = wmma16(a1, b, acc[1][t]);
        }
        __syncthreads();
        if (j + 2 < nsteps)
            async_copy_b128(cur ? lds_dst1 : lds_dst0, gsrc + (unsigned long long)(j + 2) * 64);
    }

#pragma unroll
    for (int i = 0; i < 2; i++)
#pragma unroll
        for (int t = 0; t < 4; t++)
#pragma unroll
            for (int r = 0; r < 8; r++) {
                int m = m0 + 16 * i + r + 8 * gh;     // global row (b*SEQ + s)
                int n = n0 + 16 * t + nlo;            // global col
                float val = acc[i][t][r];
                if (mode == 2) {
                    ((float*)Cv)[(long)m * N + n] = val;
                } else {
                    int bb = m >> 10, s = m & 1023;   // SEQ == 1024
                    int hh = n >> 6,  dk = n & 63;    // HDK == 64
                    long idx = (mode == 0)
                        ? ((((long)bb * NH + hh) * SEQ + s) * HDK + dk)
                        : ((((long)bb * NH + hh) * HDK + dk) * SEQ + s);
                    ((_Float16*)Cv)[idx] = (_Float16)val;
                }
            }
}

// ---------------------------------------------------------------- flash attention
// grid.x = BATCH*NH*(SEQ/128); block = 256 (8 waves); wave owns 16 query rows.
// Key tile Bc=64: 16 WMMAs per softmax pass (halves per-key softmax overhead vs Bc=32).
__global__ void __launch_bounds__(256) flash_attn_kernel(const _Float16* __restrict__ Qh,
                                                         const _Float16* __restrict__ Kh,
                                                         const _Float16* __restrict__ Vt,
                                                         const float* __restrict__ bias,
                                                         const float* __restrict__ btr,
                                                         _Float16* __restrict__ AO) {
    __shared__ _Float16 lds_p[8][16 * 72];   // per-wave 16x64 P tile, stride 72 halves (144 B, 16B-aligned)

    const int lane = threadIdx.x & 31;
    const int wave = threadIdx.x >> 5;
    const int blk  = blockIdx.x;
    const int qblk = blk & 7;
    const int bh   = blk >> 3;
    const int h = bh & 15, b = bh >> 4;
    const int q0 = qblk * 128 + wave * 16;
    const int nlo = lane & 15, gh = lane >> 4;

    const _Float16* Qbh = Qh + (((long)b * NH + h) * SEQ) * HDK;
    const _Float16* Kbh = Kh + (((long)b * NH + h) * SEQ) * HDK;
    const _Float16* Vbh = Vt + (((long)b * NH + h) * HDK) * SEQ;
    const float tb = btr[h];

    // Q fragments for this wave's 16 rows (dk 0-31 and 32-63), loaded once
    v16h aq0 = load_a_frag(Qbh, HDK, q0, 0, lane);
    v16h aq1 = load_a_frag(Qbh, HDK, q0, 32, lane);

    v8f accv[4];
#pragma unroll
    for (int t = 0; t < 4; t++) accv[t] = (v8f){0.f,0.f,0.f,0.f,0.f,0.f,0.f,0.f};
    float mrun[8], lrun[8];
#pragma unroll
    for (int r = 0; r < 8; r++) { mrun[r] = -1e30f; lrun[r] = 0.f; }

    _Float16* Pl = &lds_p[wave][0];

    for (int key0 = 0; key0 < SEQ; key0 += 64) {
        // scores: 16 queries x 64 keys, contraction over dk=64 (2 WMMAs per 16-key group)
        v8f s4[4];
#pragma unroll
        for (int kg = 0; kg < 4; kg++) {
            v16h bk0 = load_b_frag(Kbh, HDK, key0 + 16 * kg, 0, lane);
            v16h bk1 = load_b_frag(Kbh, HDK, key0 + 16 * kg, 32, lane);
            v8f z = (v8f){0.f,0.f,0.f,0.f,0.f,0.f,0.f,0.f};
            z = wmma16(aq0, bk0, z);
            z = wmma16(aq1, bk1, z);
            s4[kg] = z;
        }
        // scale + per-head lead-lag bias
#pragma unroll
        for (int r = 0; r < 8; r++) {
            int qm = q0 + r + 8 * gh;
            const float* brow = bias + (long)qm * SEQ + key0 + nlo;
#pragma unroll
            for (int kg = 0; kg < 4; kg++)
                s4[kg][r] = s4[kg][r] * 0.125f + tb * brow[16 * kg];
        }
        // online softmax: row max / rescale / exp / row sum (16-lane butterfly, half-wave safe)
#pragma unroll
        for (int r = 0; r < 8; r++) {
            float tmax = fmaxf(fmaxf(s4[0][r], s4[1][r]), fmaxf(s4[2][r], s4[3][r]));
#pragma unroll
            for (int off = 8; off >= 1; off >>= 1)
                tmax = fmaxf(tmax, __shfl_xor(tmax, off, 32));
            float mnew = fmaxf(mrun[r], tmax);
            float scale = __expf(mrun[r] - mnew);
            mrun[r] = mnew;

            float p0 = __expf(s4[0][r] - mnew);
            float p1 = __expf(s4[1][r] - mnew);
            float p2 = __expf(s4[2][r] - mnew);
            float p3 = __expf(s4[3][r] - mnew);
            float tsum = (p0 + p1) + (p2 + p3);
#pragma unroll
            for (int off = 8; off >= 1; off >>= 1)
                tsum += __shfl_xor(tsum, off, 32);
            lrun[r] = lrun[r] * scale + tsum;

            int row = r + 8 * gh;
            Pl[row * 72 +      nlo] = (_Float16)p0;
            Pl[row * 72 + 16 + nlo] = (_Float16)p1;
            Pl[row * 72 + 32 + nlo] = (_Float16)p2;
            Pl[row * 72 + 48 + nlo] = (_Float16)p3;
#pragma unroll
            for (int t = 0; t < 4; t++) accv[t][r] *= scale;
        }
        // reload P as two A fragments (16x32 each) from the per-wave LDS tile
        v16h ap0, ap1;
        {
            const _Float16* p = Pl + (long)nlo * 72 + 16 * gh;
            union { v16h v; v8h hh[2]; } u;
            u.hh[0] = *(const v8h*)(p);
            u.hh[1] = *(const v8h*)(p + 8);
            ap0 = u.v;
            u.hh[0] = *(const v8h*)(p + 32);
            u.hh[1] = *(const v8h*)(p + 40);
            ap1 = u.v;
        }
        // accv += P(16x64) * V(64x64): 8 WMMAs, B frags contiguous from transposed V
#pragma unroll
        for (int t = 0; t < 4; t++) {
            v16h bv0 = load_b_frag(Vbh, SEQ, 16 * t, key0, lane);
            v16h bv1 = load_b_frag(Vbh, SEQ, 16 * t, key0 + 32, lane);
            accv[t] = wmma16(ap0, bv0, accv[t]);
            accv[t] = wmma16(ap1, bv1, accv[t]);
        }
    }

    // normalize and store to attention output [B*S][H*DK] fp16 (A matrix of final GEMM)
#pragma unroll
    for (int r = 0; r < 8; r++) {
        float inv = 1.f / lrun[r];
        int qm = q0 + r + 8 * gh;
#pragma unroll
        for (int t = 0; t < 4; t++) {
            AO[((long)(b * SEQ + qm)) * DMODEL + h * HDK + 16 * t + nlo] =
                (_Float16)(accv[t][r] * inv);
        }
    }
}

// ---------------------------------------------------------------- launcher
extern "C" void kernel_launch(void* const* d_in, const int* in_sizes, int n_in,
                              void* d_out, int out_size, void* d_ws, size_t ws_size,
                              hipStream_t stream) {
    const float* X    = (const float*)d_in[0];
    const float* bias = (const float*)d_in[1];
    const float* Wq   = (const float*)d_in[2];
    const float* Wk   = (const float*)d_in[3];
    const float* Wv   = (const float*)d_in[4];
    const float* Wo   = (const float*)d_in[5];
    const float* btr  = (const float*)d_in[6];
    float* out = (float*)d_out;

    char* ws = (char*)d_ws;
    const long MB = 1024 * 1024;
    _Float16* Xh  = (_Float16*)(ws);              // 32 MB
    _Float16* Wqt = (_Float16*)(ws + 32 * MB);    // 2 MB each
    _Float16* Wkt = (_Float16*)(ws + 34 * MB);
    _Float16* Wvt = (_Float16*)(ws + 36 * MB);
    _Float16* Wot = (_Float16*)(ws + 38 * MB);
    _Float16* Qh  = (_Float16*)(ws + 40 * MB);    // 32 MB  [b][h][s][dk]
    _Float16* Kh  = (_Float16*)(ws + 72 * MB);    // 32 MB  [b][h][s][dk]
    _Float16* Vt  = (_Float16*)(ws + 104 * MB);   // 32 MB  [b][h][dk][s]
    _Float16* AO  = (_Float16*)(ws + 136 * MB);   // 32 MB  [B*S][H*DK]

    const long nX = (long)BATCH * SEQ * DMODEL;
    f32_to_f16_kernel<<<(int)((nX + 255) / 256), 256, 0, stream>>>(X, Xh, nX);

    const long nW = (long)DMODEL * DMODEL;
    const int wBlocks = (int)((nW + 255) / 256);
    transpose_f32_to_f16_kernel<<<wBlocks, 256, 0, stream>>>(Wq, Wqt, DMODEL, DMODEL);
    transpose_f32_to_f16_kernel<<<wBlocks, 256, 0, stream>>>(Wk, Wkt, DMODEL, DMODEL);
    transpose_f32_to_f16_kernel<<<wBlocks, 256, 0, stream>>>(Wv, Wvt, DMODEL, DMODEL);
    transpose_f32_to_f16_kernel<<<wBlocks, 256, 0, stream>>>(Wo, Wot, DMODEL, DMODEL);

    dim3 ggrid(DMODEL / 64, (BATCH * SEQ) / 256);
    gemm_wmma_kernel<<<ggrid, 256, 0, stream>>>(Xh, Wqt, Qh, BATCH * SEQ, DMODEL, DMODEL, 0);
    gemm_wmma_kernel<<<ggrid, 256, 0, stream>>>(Xh, Wkt, Kh, BATCH * SEQ, DMODEL, DMODEL, 0);
    gemm_wmma_kernel<<<ggrid, 256, 0, stream>>>(Xh, Wvt, Vt, BATCH * SEQ, DMODEL, DMODEL, 1);

    flash_attn_kernel<<<BATCH * NH * (SEQ / 128), 256, 0, stream>>>(Qh, Kh, Vt, bias, btr, AO);

    gemm_wmma_kernel<<<ggrid, 256, 0, stream>>>(AO, Wot, out, BATCH * SEQ, DMODEL, DMODEL, 2);
}
